// B_Conv2d_ConvNN_K_All_20435454394602
// MI455X (gfx1250) — matrix operand order repaired and checked by reading, not compiled, not run on hardware
//
#include <hip/hip_runtime.h>
#include <hip/hip_bf16.h>

typedef _Float16 f16;
typedef __attribute__((ext_vector_type(16))) _Float16 v16h;
typedef __attribute__((ext_vector_type(8)))  float    v8f;

// ---------------------------------------------------------------------------
// Feature probes (async global->LDS path, gfx1250)
// ---------------------------------------------------------------------------
#if defined(__has_builtin)
# if __has_builtin(__builtin_amdgcn_global_load_async_to_lds_b128)
#  define HAVE_ASYNC_COPY 1
# endif
# if __has_builtin(__builtin_amdgcn_s_wait_asynccnt)
#  define HAVE_WAIT_ASYNC_BUILTIN 1
# endif
#endif

#if defined(HAVE_ASYNC_COPY)
typedef int v4i_vs __attribute__((vector_size(16)));
typedef __attribute__((address_space(1))) v4i_vs* glob_v4i_p;
typedef __attribute__((address_space(3))) v4i_vs* lds_v4i_p;

__device__ __forceinline__ void async_copy_g2l_128(const void* g, void* l) {
  __builtin_amdgcn_global_load_async_to_lds_b128(
      (glob_v4i_p)g, (lds_v4i_p)l, /*offset=*/0, /*cpol=*/0);
}
__device__ __forceinline__ void wait_async0() {
# if defined(HAVE_WAIT_ASYNC_BUILTIN)
  __builtin_amdgcn_s_wait_asynccnt(0);
# else
  asm volatile("s_wait_asynccnt 0" ::: "memory");
# endif
}
#endif

// ---------------------------------------------------------------------------
// WMMA helpers (CDNA5 gfx1250, wave32)
// ---------------------------------------------------------------------------
__device__ __forceinline__ v8f wmma16(v16h a, v16h b, v8f c) {
  return __builtin_amdgcn_wmma_f32_16x16x32_f16(
      false, a, false, b, (short)0, c, false, false);
}

// A-fragment 16(M)x32(K), LDS tile row-major [M][K] (K contiguous, 4B aligned).
__device__ __forceinline__ v16h load_frag_A_rm(const f16* p, int mStride) {
  int lane = threadIdx.x & 31;
  int m = lane & 15;
  int h = lane >> 4;
  const unsigned* q = (const unsigned*)(p + m * mStride);
  union { v16h v; unsigned u[8]; } r;
#pragma unroll
  for (int v = 0; v < 8; ++v) {
    int kb = ((v < 4) ? 0 : 16) + h * 8 + (v & 3) * 2;  // even
    r.u[v] = q[kb >> 1];
  }
  return r.v;
}

// B-fragment 32(K)x16(N), LDS tile N-major [N][K] (K contiguous, 4B aligned).
__device__ __forceinline__ v16h load_frag_B_nm(const f16* p, int nStride) {
  int lane = threadIdx.x & 31;
  int n = lane & 15;
  int h = lane >> 4;
  const unsigned* q = (const unsigned*)(p + n * nStride);
  union { v16h v; unsigned u[8]; } r;
#pragma unroll
  for (int v = 0; v < 8; ++v) r.u[v] = q[((v < 4) ? 0 : 8) + h * 4 + (v & 3)];
  return r.v;
}

// 4x f32 (16B-aligned global) -> 4x f16 -> one 8B LDS store
__device__ __forceinline__ void cvt4_store(f16* dst, const float* src) {
  float4 v = *(const float4*)src;
  union { f16 h[4]; unsigned long long u; } r;
  r.h[0] = (f16)v.x; r.h[1] = (f16)v.y; r.h[2] = (f16)v.z; r.h[3] = (f16)v.w;
  *(unsigned long long*)dst = r.u;
}
__device__ __forceinline__ void cvt4_store_reg(f16* dst, float4 v) {
  union { f16 h[4]; unsigned long long u; } r;
  r.h[0] = (f16)v.x; r.h[1] = (f16)v.y; r.h[2] = (f16)v.z; r.h[3] = (f16)v.w;
  *(unsigned long long*)dst = r.u;
}

// register-resident top-9 insert (descending, ties keep earlier/lower index).
// Internal chain is unrolled (constant indices -> stays in VGPRs); guarded so
// the bubble only runs when some lane actually inserts.
__device__ __forceinline__ void ins9(float (&bv)[9], int (&bi)[9], float v, int j) {
  if (v > bv[8]) {
    bv[8] = v; bi[8] = j;
#pragma unroll
    for (int q = 8; q > 0; --q) {
      if (bv[q] > bv[q - 1]) {
        float tv = bv[q]; bv[q] = bv[q - 1]; bv[q - 1] = tv;
        int   ti = bi[q]; bi[q] = bi[q - 1]; bi[q - 1] = ti;
      }
    }
  }
}

// C/D layout: VGPR r, lanes 0-15 -> (M=r, N=lane); lanes 16-31 -> (M=r+8, N=lane-16)

// ---------------------------------------------------------------------------
// K0: pixel_unshuffle (s=2) + channel-L2-normalize.
// ---------------------------------------------------------------------------
__global__ void unshuffle_norm_kernel(const float* __restrict__ src, int Cin, int CPAD,
                                      float* __restrict__ U, f16* __restrict__ Xn) {
  int t = blockIdx.x * blockDim.x + threadIdx.x;
  int b = t >> 8;
  int n = t & 255;
  int h2 = n >> 4, w2 = n & 15;
  int Cu = Cin * 4;
  const float* sb = src + (size_t)b * Cin * 1024;
  float* Ub = U + (size_t)b * Cu * 256;
  f16*   Xb = Xn + ((size_t)b * 256 + n) * CPAD;
  float ss = 0.f;
  for (int cu = 0; cu < Cu; ++cu) {
    int c = cu >> 2, s1 = (cu >> 1) & 1, s2 = cu & 1;
    float v = sb[(size_t)c * 1024 + (h2 * 2 + s1) * 32 + (w2 * 2 + s2)];
    Ub[cu * 256 + n] = v;
    ss += v * v;
  }
  float inv = 1.f / (sqrtf(ss) + 1e-8f);
  for (int c = 0; c < CPAD; ++c)
    Xb[c] = (c < Cu) ? (f16)(Ub[c * 256 + n] * inv) : (f16)0.f;
}

// ---------------------------------------------------------------------------
// K1: cosine-similarity Gram tile (WMMA) + fused parallel top-9.
// Stage is a contiguous byte copy -> GLOBAL_LOAD_ASYNC_TO_LDS_B128.
// grid: B*16 blocks (16 sim rows each), 128 threads (4 waves).
// ---------------------------------------------------------------------------
template <int CU, int CPAD>
__global__ void sim_topk_kernel(const f16* __restrict__ Xn, int* __restrict__ idx) {
  __shared__ __align__(16) f16   XsT[256][CPAD];  // pixel-major normalized features
  __shared__ __align__(16) float simL[16][256];
  __shared__ float candV[16][8][9];
  __shared__ int   candI[16][8][9];

  int b  = blockIdx.x >> 4;
  int m0 = (blockIdx.x & 15) * 16;
  const f16* Xb = Xn + (size_t)b * 256 * CPAD;
  constexpr int BYTES = 256 * CPAD * 2;

#if defined(HAVE_ASYNC_COPY)
  for (int off = (int)threadIdx.x * 16; off < BYTES; off += (int)blockDim.x * 16)
    async_copy_g2l_128((const char*)Xb + off, (char*)&XsT[0][0] + off);
  wait_async0();
#else
  {
    const uint4* gs = (const uint4*)Xb;
    uint4* ls = (uint4*)&XsT[0][0];
    for (int i = threadIdx.x; i < BYTES / 16; i += blockDim.x) ls[i] = gs[i];
  }
#endif
  __syncthreads();

  int wave = threadIdx.x >> 5;
  int lane = threadIdx.x & 31;
#pragma unroll
  for (int t = 0; t < 4; ++t) {
    int j0 = wave * 64 + t * 16;
    v8f acc = {};
#pragma unroll
    for (int kk = 0; kk < CPAD; kk += 32) {
      v16h a  = load_frag_A_rm(&XsT[m0][kk], CPAD);
      v16h bb = load_frag_B_nm(&XsT[j0][kk], CPAD);
      acc = wmma16(a, bb, acc);
    }
    int n = lane & 15, mh = (lane >> 4) << 3;
#pragma unroll
    for (int r = 0; r < 8; ++r) simL[r + mh][j0 + n] = acc[r];
  }
  __syncthreads();

  // phase 1: 8 threads per row, 32-element segments, register top-9
  {
    int row = (int)threadIdx.x >> 3;
    int seg = (int)threadIdx.x & 7;
    float bv[9];
    int   bi[9];
#pragma unroll
    for (int q = 0; q < 9; ++q) { bv[q] = -3.0e38f; bi[q] = 0; }
    int j0 = seg * 32;
#pragma unroll 1
    for (int jj = 0; jj < 32; ++jj) ins9(bv, bi, simL[row][j0 + jj], j0 + jj);
#pragma unroll
    for (int q = 0; q < 9; ++q) { candV[row][seg][q] = bv[q]; candI[row][seg][q] = bi[q]; }
  }
  __syncthreads();

  // phase 2: 16 threads merge 8x9 candidates per row
  if (threadIdx.x < 16) {
    int row = threadIdx.x;
    float bv[9];
    int   bi[9];
#pragma unroll
    for (int q = 0; q < 9; ++q) { bv[q] = -3.0e38f; bi[q] = 0; }
#pragma unroll 1
    for (int s = 0; s < 8; ++s)
#pragma unroll 1
      for (int q = 0; q < 9; ++q) ins9(bv, bi, candV[row][s][q], candI[row][s][q]);
    int* op = idx + ((size_t)b * 256 + (m0 + row)) * 9;
#pragma unroll
    for (int q = 0; q < 9; ++q) op[q] = bi[q];
  }
}

// ---------------------------------------------------------------------------
// K2: branch GEMM (implicit im2col): 3x3 conv (zero-pad SAME) or KNN gather.
// ---------------------------------------------------------------------------
template <int CU, int COUT, int KPAD, bool KNN>
__global__ void branch_gemm_kernel(const float* __restrict__ U,
                                   const float* __restrict__ W,
                                   const float* __restrict__ bias,
                                   const int* __restrict__ idx,
                                   float* __restrict__ cat, int catC, int coff) {
  constexpr int KREAL = CU * 9;
  constexpr int KR4   = KREAL / 4;   // KREAL is a multiple of 4 (108, 576)
  __shared__ __align__(16) f16 As[16][KPAD];
  __shared__ __align__(16) f16 Ws[COUT][KPAD];   // N-major

  int b  = blockIdx.x >> 4;
  int n0 = (blockIdx.x & 15) * 16;
  const float* Ub = U + (size_t)b * CU * 256;

  // stage weights: vectorized f32->f16 (16B-aligned: KREAL*4 % 16 == 0)
  for (int e4 = threadIdx.x; e4 < COUT * KR4; e4 += blockDim.x) {
    int o = e4 / KR4, k4 = (e4 % KR4) * 4;
    cvt4_store(&Ws[o][k4], &W[(size_t)o * KREAL + k4]);
  }
  if constexpr (KPAD > KREAL) {
    for (int e = threadIdx.x; e < COUT * (KPAD - KREAL); e += blockDim.x) {
      int o = e / (KPAD - KREAL), k = KREAL + e % (KPAD - KREAL);
      Ws[o][k] = (f16)0.f;
    }
  }
  // stage A (gather / im2col) - inherently scalar
  for (int e = threadIdx.x; e < 16 * KPAD; e += blockDim.x) {
    int m = e / KPAD, k = e % KPAD;
    float v = 0.f;
    if (k < KREAL) {
      int c = k / 9, j = k % 9;
      int n = n0 + m;
      int p;
      if constexpr (KNN) {
        p = idx[((size_t)b * 256 + n) * 9 + j];
      } else {
        int y = n >> 4, x = n & 15;
        int yy = y + j / 3 - 1, xx = x + j % 3 - 1;
        p = (yy >= 0 && yy < 16 && xx >= 0 && xx < 16) ? (yy * 16 + xx) : -1;
      }
      if (p >= 0) v = Ub[c * 256 + p];
    }
    As[m][k] = (f16)v;
  }
  __syncthreads();

  int wave = threadIdx.x >> 5;
  int lane = threadIdx.x & 31;
  int o0 = wave * 16;
  v8f acc;
  float bv = bias[o0 + (lane & 15)];
#pragma unroll
  for (int r = 0; r < 8; ++r) acc[r] = bv;
#pragma unroll
  for (int kk = 0; kk < KPAD; kk += 32) {
    v16h a  = load_frag_A_rm(&As[0][kk], KPAD);
    v16h bb = load_frag_B_nm(&Ws[o0][kk], KPAD);
    acc = wmma16(a, bb, acc);
  }
  int n = lane & 15, mh = (lane >> 4) << 3;
  float* Cb = cat + (size_t)b * catC * 256;
#pragma unroll
  for (int r = 0; r < 8; ++r) {
    int m = r + mh;
    float v = acc[r];
    v = v > 0.f ? v : 0.f;
    Cb[(size_t)(coff + o0 + n) * 256 + n0 + m] = v;
  }
}

// ---------------------------------------------------------------------------
// K3: 1x1 combine conv + ReLU + fused pixel_shuffle (s=2) scatter.
// ---------------------------------------------------------------------------
template <int C2, int COUT2, int CNEXT>
__global__ void combine_shuffle_kernel(const float* __restrict__ cat,
                                       const float* __restrict__ Wp,
                                       const float* __restrict__ bias,
                                       float* __restrict__ hout) {
  __shared__ __align__(16) f16 As[16][C2];
  __shared__ __align__(16) f16 Ws[COUT2][C2];    // N-major == native Wp layout

  int b  = blockIdx.x >> 4;
  int n0 = (blockIdx.x & 15) * 16;
  const float* Cb = cat + (size_t)b * C2 * 256;

  // A stage: pixel index fastest -> coalesced global reads
  for (int e = threadIdx.x; e < 16 * C2; e += blockDim.x) {
    int m = e & 15, k = e >> 4;
    As[m][k] = (f16)Cb[(size_t)k * 256 + n0 + m];
  }
  // weights: contiguous, vectorized
  for (int e4 = threadIdx.x; e4 < COUT2 * C2 / 4; e4 += blockDim.x)
    cvt4_store(&Ws[0][0] + e4 * 4, Wp + e4 * 4);
  __syncthreads();

  int wave = threadIdx.x >> 5;
  int lane = threadIdx.x & 31;
  constexpr int TPW = COUT2 / 64;
#pragma unroll
  for (int t = 0; t < TPW; ++t) {
    int o0 = (wave * TPW + t) * 16;
    v8f acc;
    float bv = bias[o0 + (lane & 15)];
#pragma unroll
    for (int r = 0; r < 8; ++r) acc[r] = bv;
#pragma unroll
    for (int kk = 0; kk < C2; kk += 32) {
      v16h a  = load_frag_A_rm(&As[0][kk], C2);
      v16h bb = load_frag_B_nm(&Ws[o0][kk], C2);
      acc = wmma16(a, bb, acc);
    }
    int n = lane & 15, mh = (lane >> 4) << 3;
#pragma unroll
    for (int r = 0; r < 8; ++r) {
      int m = r + mh;
      int o = o0 + n;
      float v = acc[r];
      v = v > 0.f ? v : 0.f;
      int pix = n0 + m, y = pix >> 4, x = pix & 15;
      int c = o >> 2, s1 = (o >> 1) & 1, s2 = o & 1;
      hout[(((size_t)b * CNEXT + c) * 32 + (2 * y + s1)) * 32 + (2 * x + s2)] = v;
    }
  }
}

// ---------------------------------------------------------------------------
// K4: FC1  F(256,1024) = relu( H(256,32768) * W1^T + b ).  HBM-bound on W1.
// grid: (16 Nblocks x 4 Mblocks); 128 threads (4 waves); k-chunk = 64.
// Software-pipelined: next chunk's 16 float4 loads overlap current 8 WMMAs.
// ---------------------------------------------------------------------------
__global__ void fc1_kernel(const float* __restrict__ H, const float* __restrict__ W1,
                           const float* __restrict__ b1, float* __restrict__ F) {
  __shared__ __align__(16) f16 As[64][64];   // row-major   (8 KB)
  __shared__ __align__(16) f16 Bs[64][64];   // N-major     (8 KB)
  const int tid  = (int)threadIdx.x & 127;   // provable bound -> no exec guards
  const int n0   = blockIdx.x * 64;
  const int m0   = blockIdx.y * 64;
  const int wave = tid >> 5;
  const int lane = tid & 31;
  const int rowq = tid >> 4;        // 0..7
  const int kq   = (tid & 15) * 4;  // 0..60

  v8f acc[4];
#pragma unroll
  for (int t = 0; t < 4; ++t) {
    float bv = b1[n0 + t * 16 + (lane & 15)];
#pragma unroll
    for (int r = 0; r < 8; ++r) acc[t][r] = bv;
  }

  const float* Hrow = H  + (size_t)m0 * 32768;
  const float* Wrow = W1 + (size_t)n0 * 32768;

  float4 ar[8], br[8];
  auto load_regs = [&](int kb) {
#pragma unroll
    for (int i = 0; i < 8; ++i)
      ar[i] = *(const float4*)&Hrow[(size_t)(i * 8 + rowq) * 32768 + kb + kq];
#pragma unroll
    for (int i = 0; i < 8; ++i)
      br[i] = *(const float4*)&Wrow[(size_t)(i * 8 + rowq) * 32768 + kb + kq];
  };
  auto store_lds = [&]() {
#pragma unroll
    for (int i = 0; i < 8; ++i) cvt4_store_reg(&As[i * 8 + rowq][kq], ar[i]);
#pragma unroll
    for (int i = 0; i < 8; ++i) cvt4_store_reg(&Bs[i * 8 + rowq][kq], br[i]);
  };

  load_regs(0);
  for (int kb = 0; kb < 32768; kb += 64) {
    store_lds();
    __syncthreads();
    if (kb + 64 < 32768) load_regs(kb + 64);   // in flight during WMMAs
    if (kb + 128 < 32768)
      __builtin_prefetch(&Wrow[(size_t)rowq * 32768 + kb + 128], 0, 1);
#pragma unroll
    for (int ks = 0; ks < 64; ks += 32) {
      v16h a = load_frag_A_rm(&As[wave * 16][ks], 64);
#pragma unroll
      for (int t = 0; t < 4; ++t) {
        v16h bb = load_frag_B_nm(&Bs[t * 16][ks], 64);
        acc[t] = wmma16(a, bb, acc[t]);
      }
    }
    __syncthreads();
  }

  int n = lane & 15, mh = (lane >> 4) << 3;
#pragma unroll
  for (int t = 0; t < 4; ++t)
#pragma unroll
    for (int r = 0; r < 8; ++r) {
      float v = acc[t][r];
      v = v > 0.f ? v : 0.f;
      F[(size_t)(m0 + wave * 16 + r + mh) * 1024 + n0 + t * 16 + n] = v;
    }
}

// ---------------------------------------------------------------------------
// K5: FC2 (256x10, K=1024) — tiny, plain VALU.
// ---------------------------------------------------------------------------
__global__ void fc2_kernel(const float* __restrict__ F, const float* __restrict__ W2,
                           const float* __restrict__ b2, float* __restrict__ out) {
  int i = blockIdx.x * blockDim.x + threadIdx.x;
  if (i >= 2560) return;
  int row = i / 10, col = i % 10;
  const float* f = F + (size_t)row * 1024;
  const float* w = W2 + (size_t)col * 1024;
  float s = b2[col];
  for (int k = 0; k < 1024; ++k) s += f[k] * w[k];
  out[i] = s;
}

// ---------------------------------------------------------------------------
// Host-side orchestration
// ---------------------------------------------------------------------------
extern "C" void kernel_launch(void* const* d_in, const int* in_sizes, int n_in,
                              void* d_out, int out_size, void* d_ws, size_t ws_size,
                              hipStream_t stream) {
  (void)in_sizes; (void)n_in; (void)out_size; (void)ws_size;
  const float* x    = (const float*)d_in[0];
  const float* Wc1  = (const float*)d_in[1];
  const float* bc1  = (const float*)d_in[2];
  const float* Wn1  = (const float*)d_in[3];
  const float* bn1  = (const float*)d_in[4];
  const float* Wp1  = (const float*)d_in[5];
  const float* bp1  = (const float*)d_in[6];
  const float* Wc2  = (const float*)d_in[7];
  const float* bc2  = (const float*)d_in[8];
  const float* Wn2  = (const float*)d_in[9];
  const float* bn2  = (const float*)d_in[10];
  const float* Wp2  = (const float*)d_in[11];
  const float* bp2  = (const float*)d_in[12];
  const float* Wfc1 = (const float*)d_in[13];
  const float* bfc1 = (const float*)d_in[14];
  const float* Wfc2 = (const float*)d_in[15];
  const float* bfc2 = (const float*)d_in[16];

  char* ws = (char*)d_ws;
  float* U    = (float*)(ws + 0);            // 256*64*256*4  = 16 MiB
  f16*   Xn   = (f16*)  (ws + 16777216);     // 256*256*64*2  =  8 MiB
  int*   idx  = (int*)  (ws + 25165824);     // 256*256*9*4   = 2.25 MiB
  float* cat  = (float*)(ws + 27525120);     // 256*64*256*4  = 16 MiB
  float* h1   = (float*)(ws + 44302336);     // 256*16*1024*4 = 16 MiB
  float* h2   = (float*)(ws + 61079552);     // 256*32*1024*4 = 32 MiB
  float* F    = (float*)(ws + 94633984);     // 256*1024*4    =  1 MiB

  const int B = 256;

  // ------------------------- Block 1 (Cu=12, Cout=16) ----------------------
  unshuffle_norm_kernel<<<B, 256, 0, stream>>>(x, 3, 32, U, Xn);
  sim_topk_kernel<12, 32><<<B * 16, 128, 0, stream>>>(Xn, idx);
  branch_gemm_kernel<12, 16, 128, false><<<B * 16, 32, 0, stream>>>(
      U, Wc1, bc1, nullptr, cat, 32, 0);
  branch_gemm_kernel<12, 16, 128, true><<<B * 16, 32, 0, stream>>>(
      U, Wn1, bn1, idx, cat, 32, 16);
  combine_shuffle_kernel<32, 64, 16><<<B * 16, 128, 0, stream>>>(cat, Wp1, bp1, h1);

  // ------------------------- Block 2 (Cu=64, Cout=32) ----------------------
  unshuffle_norm_kernel<<<B, 256, 0, stream>>>(h1, 16, 64, U, Xn);
  sim_topk_kernel<64, 64><<<B * 16, 128, 0, stream>>>(Xn, idx);
  branch_gemm_kernel<64, 32, 576, false><<<B * 16, 64, 0, stream>>>(
      U, Wc2, bc2, nullptr, cat, 64, 0);
  branch_gemm_kernel<64, 32, 576, true><<<B * 16, 64, 0, stream>>>(
      U, Wn2, bn2, idx, cat, 64, 32);
  combine_shuffle_kernel<64, 128, 32><<<B * 16, 128, 0, stream>>>(cat, Wp2, bp2, h2);

  // ------------------------------ FC head ----------------------------------
  fc1_kernel<<<dim3(16, 4), 128, 0, stream>>>(h2, Wfc1, bfc1, F);
  fc2_kernel<<<10, 256, 0, stream>>>(F, Wfc2, bfc2, (float*)d_out);
}
